// SelfAttention_30236569763908
// MI455X (gfx1250) — matrix operand order repaired and checked by reading, not compiled, not run on hardware
//
#include <hip/hip_runtime.h>
#include <hip/hip_bf16.h>
#include <math.h>

typedef __attribute__((ext_vector_type(16))) __bf16 v16bf;
typedef __attribute__((ext_vector_type(8)))  __bf16 v8bf;
typedef __attribute__((ext_vector_type(8)))  float  v8f;

#define D_MODEL  1024
#define N_HEAD   16
#define HEAD_DIM 64
#define BATCH    2
#define SEQ      2048
#define M_TOK    (BATCH * SEQ)       // 4096 tokens

// ---------------------------------------------------------------------------
// helpers
// ---------------------------------------------------------------------------
__device__ __forceinline__ v16bf combine8(v8bf lo, v8bf hi) {
  v16bf r;
#pragma unroll
  for (int i = 0; i < 8; ++i) { r[i] = lo[i]; r[i + 8] = hi[i]; }
  return r;
}

__device__ __forceinline__ v8f wmma_bf16(v16bf a, v16bf b, v8f c) {
  return __builtin_amdgcn_wmma_f32_16x16x32_bf16(false, a, false, b, (short)0, c,
                                                 false, false);
}

// A-fragment (16x32, M = lane%16): two 16B loads at K offsets {8*half, 16+8*half}
__device__ __forceinline__ v16bf load_afrag(const __bf16* row, int k0, int half) {
  return combine8(*(const v8bf*)(row + k0 + 8 * half),
                  *(const v8bf*)(row + k0 + 16 + 8 * half));
}
// B-fragment (32x16, N = lane%16): two 16B loads at K offsets {8*half, 16+8*half}
__device__ __forceinline__ v16bf load_bfrag(const __bf16* row, int k0, int half) {
  return combine8(*(const v8bf*)(row + k0 + 8 * half),
                  *(const v8bf*)(row + k0 + 16 + 8 * half));
}

// ---------------------------------------------------------------------------
// f32 -> bf16 cast
// ---------------------------------------------------------------------------
__global__ void cast_f32_bf16(const float* __restrict__ in,
                              __bf16* __restrict__ out, int n) {
  int i = blockIdx.x * blockDim.x + threadIdx.x;
  int stride = gridDim.x * blockDim.x;
  for (; i < n; i += stride) out[i] = (__bf16)in[i];
}

// ---------------------------------------------------------------------------
// Register-blocked GEMM core: one wave computes a 32x64 block of
// C = A(MxK, row-major) * W(NxK, row-major)^T.
// acc[i][c] is the 16x16 D-tile at rows m0+16*i, cols n0+16*c.
// ---------------------------------------------------------------------------
__device__ __forceinline__ void gemm_block_bf16(const __bf16* __restrict__ A,
                                                const __bf16* __restrict__ W,
                                                int K, int m0, int n0,
                                                v8f acc[2][4]) {
  const int lane = threadIdx.x & 31;
  const int half = lane >> 4;
  const int lm   = lane & 15;
  v8f zero = {0.f, 0.f, 0.f, 0.f, 0.f, 0.f, 0.f, 0.f};
#pragma unroll
  for (int i = 0; i < 2; ++i)
#pragma unroll
    for (int c = 0; c < 4; ++c) acc[i][c] = zero;

  const __bf16* arow0 = A + (size_t)(m0 + lm) * K;
  const __bf16* arow1 = A + (size_t)(m0 + 16 + lm) * K;
  const __bf16* brow0 = W + (size_t)(n0 + lm) * K;
  const __bf16* brow1 = W + (size_t)(n0 + 16 + lm) * K;
  const __bf16* brow2 = W + (size_t)(n0 + 32 + lm) * K;
  const __bf16* brow3 = W + (size_t)(n0 + 48 + lm) * K;

  for (int k0 = 0; k0 < K; k0 += 32) {
    v16bf a0 = load_afrag(arow0, k0, half);
    v16bf a1 = load_afrag(arow1, k0, half);
    v16bf b0 = load_bfrag(brow0, k0, half);
    v16bf b1 = load_bfrag(brow1, k0, half);
    v16bf b2 = load_bfrag(brow2, k0, half);
    v16bf b3 = load_bfrag(brow3, k0, half);
    acc[0][0] = wmma_bf16(a0, b0, acc[0][0]);
    acc[0][1] = wmma_bf16(a0, b1, acc[0][1]);
    acc[0][2] = wmma_bf16(a0, b2, acc[0][2]);
    acc[0][3] = wmma_bf16(a0, b3, acc[0][3]);
    acc[1][0] = wmma_bf16(a1, b0, acc[1][0]);
    acc[1][1] = wmma_bf16(a1, b1, acc[1][1]);
    acc[1][2] = wmma_bf16(a1, b2, acc[1][2]);
    acc[1][3] = wmma_bf16(a1, b3, acc[1][3]);
  }
}

// ---------------------------------------------------------------------------
// QKV projection: qkv = x @ Wqkv^T + b ; scatter to Q[B,H,L,64] K[B,H,L,64]
// and V transposed Vt[B,H,64,L]. N-block of 64 aligns exactly with one head.
// ---------------------------------------------------------------------------
__global__ void qkv_gemm_kernel(const __bf16* __restrict__ xb,
                                const __bf16* __restrict__ w,
                                const float*  __restrict__ bias,
                                __bf16* __restrict__ Qb,
                                __bf16* __restrict__ Kb,
                                __bf16* __restrict__ Vt) {
  const int NT_N = (3 * D_MODEL) / 64;  // 48 blocks along N
  int tile = blockIdx.x * (blockDim.x >> 5) + (threadIdx.x >> 5);
  int m0 = (tile / NT_N) * 32;
  int n0 = (tile % NT_N) * 64;

  v8f acc[2][4];
  gemm_block_bf16(xb, w, D_MODEL, m0, n0, acc);

  const int lane = threadIdx.x & 31, half = lane >> 4, lm = lane & 15;
  int sec = n0 >> 10;                 // 0:Q 1:K 2:V (64-block never crosses)
  int h   = (n0 & (D_MODEL - 1)) >> 6;  // one head per 64-block
  float bn[4];
#pragma unroll
  for (int c = 0; c < 4; ++c) bn[c] = bias[n0 + c * 16 + lm];

#pragma unroll
  for (int i = 0; i < 2; ++i) {
#pragma unroll
    for (int r = 0; r < 8; ++r) {
      int t = m0 + 16 * i + r + 8 * half;   // token
      int b = t >> 11, l = t & (SEQ - 1);
#pragma unroll
      for (int c = 0; c < 4; ++c) {
        int dd = c * 16 + lm;               // head-dim index
        __bf16 bv = (__bf16)(acc[i][c][r] + bn[c]);
        if (sec == 0)
          Qb[(((size_t)b * N_HEAD + h) * SEQ + l) * HEAD_DIM + dd] = bv;
        else if (sec == 1)
          Kb[(((size_t)b * N_HEAD + h) * SEQ + l) * HEAD_DIM + dd] = bv;
        else
          Vt[(((size_t)b * N_HEAD + h) * HEAD_DIM + dd) * SEQ + l] = bv;
      }
    }
  }
}

// ---------------------------------------------------------------------------
// Causal flash attention, one wave per (b,h,32-query tile).
// Two 16-query groups share every K-tile A-fragment and V-tile B-fragment.
// Scores computed transposed (St = K * Q^T) so exp'd P lands directly in the
// A-fragment layout for the P*V WMMA.
// ---------------------------------------------------------------------------
__global__ void attn_kernel(const __bf16* __restrict__ Qb,
                            const __bf16* __restrict__ Kb,
                            const __bf16* __restrict__ Vt,
                            __bf16* __restrict__ ctx) {
  const int lane = threadIdx.x & 31, half = lane >> 4, lm = lane & 15;
  int tile = blockIdx.x * (blockDim.x >> 5) + (threadIdx.x >> 5);
  const int QT = SEQ / 32;  // 64 query tiles per (b,h)
  int bh = tile / QT;
  int q0 = (tile % QT) * 32;
  int b = bh / N_HEAD, h = bh % N_HEAD;

  const __bf16* Qbase = Qb + (size_t)bh * SEQ * HEAD_DIM;
  const __bf16* Kbase = Kb + (size_t)bh * SEQ * HEAD_DIM;
  const __bf16* Vbase = Vt + (size_t)bh * HEAD_DIM * SEQ;

  // Q as B-fragments (N = query = lane%16); per group, two d-chunks of 32
  v16bf bq[2][2];
#pragma unroll
  for (int g = 0; g < 2; ++g) {
    const __bf16* qrow = Qbase + (size_t)(q0 + 16 * g + lm) * HEAD_DIM;
    bq[g][0] = load_bfrag(qrow, 0, half);
    bq[g][1] = load_bfrag(qrow, 32, half);
  }

  v8f zero = {0.f, 0.f, 0.f, 0.f, 0.f, 0.f, 0.f, 0.f};
  v8f o[2][4];
#pragma unroll
  for (int g = 0; g < 2; ++g)
#pragma unroll
    for (int c = 0; c < 4; ++c) o[g][c] = zero;
  float m[2]    = {-INFINITY, -INFINITY};
  float lsum[2] = {0.f, 0.f};
  const float scale = 0.125f;  // 1/sqrt(64)

  for (int k0 = 0; k0 <= q0 + 31; k0 += 32) {
    // K-tile A-fragments (shared by both query groups)
    const __bf16* krow0 = Kbase + (size_t)(k0 + lm) * HEAD_DIM;
    const __bf16* krow1 = Kbase + (size_t)(k0 + 16 + lm) * HEAD_DIM;
    v16bf ka00 = load_afrag(krow0, 0, half);   // keys k0..k0+15, d 0..31
    v16bf ka01 = load_afrag(krow0, 32, half);  // keys k0..k0+15, d 32..63
    v16bf ka10 = load_afrag(krow1, 0, half);   // keys k0+16..k0+31, d 0..31
    v16bf ka11 = load_afrag(krow1, 32, half);  // keys k0+16..k0+31, d 32..63

    v16bf pf[2];
    float al[2][8];
#pragma unroll
    for (int g = 0; g < 2; ++g) {
      v8f s0 = zero, s1 = zero;
      s0 = wmma_bf16(ka00, bq[g][0], s0);
      s0 = wmma_bf16(ka01, bq[g][1], s0);
      s1 = wmma_bf16(ka10, bq[g][0], s1);
      s1 = wmma_bf16(ka11, bq[g][1], s1);

      const int q = q0 + 16 * g + lm;  // this lane's query index (group g)
      float p0[8], p1[8];
      float rm = -INFINITY;
#pragma unroll
      for (int r = 0; r < 8; ++r) {
        int key0 = k0 + r + 8 * half;
        float v0 = s0[r] * scale; if (key0 > q)      v0 = -INFINITY;
        float v1 = s1[r] * scale; if (key0 + 16 > q) v1 = -INFINITY;
        p0[r] = v0; p1[r] = v1;
        rm = fmaxf(rm, fmaxf(v0, v1));
      }
      rm = fmaxf(rm, __shfl_xor(rm, 16, 32));  // both key-halves of this query
      float mnew  = fmaxf(m[g], rm);
      float alpha = __expf(m[g] - mnew);
      float rs = 0.f;
#pragma unroll
      for (int r = 0; r < 8; ++r) {
        float e0 = __expf(p0[r] - mnew);
        float e1 = __expf(p1[r] - mnew);
        rs += e0 + e1;
        pf[g][r]     = (__bf16)e0;  // A elem e<8  -> key = e + 8*half
        pf[g][r + 8] = (__bf16)e1;  // A elem e>=8 -> key = 16 + (e-8) + 8*half
      }
      rs += __shfl_xor(rs, 16, 32);
      lsum[g] = lsum[g] * alpha + rs;
      m[g] = mnew;
      // broadcast alpha to the O row layout (row = r + 8*half across VGPRs)
#pragma unroll
      for (int r = 0; r < 8; ++r) al[g][r] = __shfl(alpha, r + 8 * half, 32);
    }

    // O += P * V : V-tile B-fragments shared by both query groups
#pragma unroll
    for (int c = 0; c < 4; ++c) {
      const __bf16* vrow = Vbase + (size_t)(c * 16 + lm) * SEQ;
      v16bf bv = load_bfrag(vrow, k0, half);
#pragma unroll
      for (int g = 0; g < 2; ++g) {
#pragma unroll
        for (int r = 0; r < 8; ++r) o[g][c][r] *= al[g][r];
        o[g][c] = wmma_bf16(pf[g], bv, o[g][c]);
      }
    }
  }

#pragma unroll
  for (int g = 0; g < 2; ++g) {
    float li[8];
#pragma unroll
    for (int r = 0; r < 8; ++r) li[r] = __shfl(lsum[g], r + 8 * half, 32);
#pragma unroll
    for (int r = 0; r < 8; ++r) {
      int t = q0 + 16 * g + r + 8 * half;
      size_t base = ((size_t)b * SEQ + t) * D_MODEL + h * HEAD_DIM;
      ctx[base +  0 + lm] = (__bf16)(o[g][0][r] / li[r]);
      ctx[base + 16 + lm] = (__bf16)(o[g][1][r] / li[r]);
      ctx[base + 32 + lm] = (__bf16)(o[g][2][r] / li[r]);
      ctx[base + 48 + lm] = (__bf16)(o[g][3][r] / li[r]);
    }
  }
}

// ---------------------------------------------------------------------------
// Output projection: out = ctx @ Wout^T + b (fp32 result)
// ---------------------------------------------------------------------------
__global__ void out_gemm_kernel(const __bf16* __restrict__ ctx,
                                const __bf16* __restrict__ w,
                                const float*  __restrict__ bias,
                                float* __restrict__ out) {
  const int NT_N = D_MODEL / 64;  // 16 blocks along N
  int tile = blockIdx.x * (blockDim.x >> 5) + (threadIdx.x >> 5);
  int m0 = (tile / NT_N) * 32;
  int n0 = (tile % NT_N) * 64;

  v8f acc[2][4];
  gemm_block_bf16(ctx, w, D_MODEL, m0, n0, acc);

  const int lane = threadIdx.x & 31, half = lane >> 4, lm = lane & 15;
  float bn[4];
#pragma unroll
  for (int c = 0; c < 4; ++c) bn[c] = bias[n0 + c * 16 + lm];
#pragma unroll
  for (int i = 0; i < 2; ++i) {
#pragma unroll
    for (int r = 0; r < 8; ++r) {
      int t = m0 + 16 * i + r + 8 * half;
#pragma unroll
      for (int c = 0; c < 4; ++c)
        out[(size_t)t * D_MODEL + n0 + c * 16 + lm] = acc[i][c][r] + bn[c];
    }
  }
}

// ---------------------------------------------------------------------------
// launch
// ---------------------------------------------------------------------------
extern "C" void kernel_launch(void* const* d_in, const int* in_sizes, int n_in,
                              void* d_out, int out_size, void* d_ws, size_t ws_size,
                              hipStream_t stream) {
  const float* x      = (const float*)d_in[0];
  const float* w_qkv  = (const float*)d_in[1];
  const float* b_qkv  = (const float*)d_in[2];
  const float* w_out  = (const float*)d_in[3];
  const float* b_out  = (const float*)d_in[4];
  float* out = (float*)d_out;

  // workspace layout (bytes)
  char* base = (char*)d_ws;
  const size_t SZ_XB   = (size_t)M_TOK * D_MODEL * 2;          //  8 MB
  const size_t SZ_WQKV = (size_t)3 * D_MODEL * D_MODEL * 2;    //  6 MB
  const size_t SZ_WO   = (size_t)D_MODEL * D_MODEL * 2;        //  2 MB
  const size_t SZ_HEAD = (size_t)BATCH * N_HEAD * SEQ * HEAD_DIM * 2;  // 8 MB each
  __bf16* xb   = (__bf16*)(base);
  __bf16* wqkv = (__bf16*)(base + SZ_XB);
  __bf16* wo   = (__bf16*)(base + SZ_XB + SZ_WQKV);
  __bf16* Qb   = (__bf16*)(base + SZ_XB + SZ_WQKV + SZ_WO);
  __bf16* Kb   = (__bf16*)(base + SZ_XB + SZ_WQKV + SZ_WO + SZ_HEAD);
  __bf16* Vt   = (__bf16*)(base + SZ_XB + SZ_WQKV + SZ_WO + 2 * SZ_HEAD);
  __bf16* ctx  = (__bf16*)(base + SZ_XB + SZ_WQKV + SZ_WO + 3 * SZ_HEAD);

  // casts
  {
    int n = M_TOK * D_MODEL;
    cast_f32_bf16<<<(n + 255) / 256, 256, 0, stream>>>(x, xb, n);
    n = 3 * D_MODEL * D_MODEL;
    cast_f32_bf16<<<(n + 255) / 256, 256, 0, stream>>>(w_qkv, wqkv, n);
    n = D_MODEL * D_MODEL;
    cast_f32_bf16<<<(n + 255) / 256, 256, 0, stream>>>(w_out, wo, n);
  }

  // QKV projection: (4096/32) x (3072/64) = 128 x 48 = 6144 waves
  {
    int tiles = (M_TOK / 32) * ((3 * D_MODEL) / 64);
    qkv_gemm_kernel<<<tiles / 8, 256, 0, stream>>>(xb, wqkv, b_qkv, Qb, Kb, Vt);
  }

  // attention: B*H*(L/32) = 2048 waves
  {
    int tiles = BATCH * N_HEAD * (SEQ / 32);
    attn_kernel<<<tiles / 8, 256, 0, stream>>>(Qb, Kb, Vt, ctx);
  }

  // output projection: (4096/32) x (1024/64) = 128 x 16 = 2048 waves
  {
    int tiles = (M_TOK / 32) * (D_MODEL / 64);
    out_gemm_kernel<<<tiles / 8, 256, 0, stream>>>(ctx, wo, b_out, out);
  }
}
